// RelationalAttention_68401649156550
// MI455X (gfx1250) — matrix operand order repaired and checked by reading, not compiled, not run on hardware
//
#include <hip/hip_runtime.h>

typedef __attribute__((ext_vector_type(16))) _Float16 v16h;
typedef __attribute__((ext_vector_type(8)))  _Float16 v8h;
typedef __attribute__((ext_vector_type(8)))  float    v8f;

#define B_  4
#define L_  1024
#define D_  1024
#define NH  16
#define HD  64
#define NR  16
#define RD  64
#define BL  (B_ * L_)
#define ATTN_SCALE 0.125f
#define REL_SCALE  0.125f

__device__ __forceinline__ v8f wmma16(v16h a, v16h b, v8f c) {
  return __builtin_amdgcn_wmma_f32_16x16x32_f16(false, a, false, b, (short)0, c, false, false);
}

// ---------------------------------------------------------------- converts --
__global__ void k_cvt_f16(const float* __restrict__ in, _Float16* __restrict__ out, int n) {
  int i = blockIdx.x * blockDim.x + threadIdx.x;
  int stride = gridDim.x * blockDim.x;
  for (; i < n; i += stride) out[i] = (_Float16)in[i];
}

// weights: in f32 [K,N] -> out f16 [N,K] (so WMMA B frags read contiguous K)
__global__ void k_wtrans(const float* __restrict__ in, _Float16* __restrict__ out, int K, int N) {
  __shared__ float tile[32][33];
  int kx = blockIdx.x * 32, nx = blockIdx.y * 32;
  for (int r = threadIdx.y; r < 32; r += 8)
    tile[r][threadIdx.x] = in[(size_t)(kx + r) * N + nx + threadIdx.x];
  __syncthreads();
  for (int r = threadIdx.y; r < 32; r += 8)
    out[(size_t)(nx + r) * K + kx + threadIdx.x] = (_Float16)tile[threadIdx.x][r];
}

// per-batch f16 transpose: in [b][R][C] -> out [b][C][R]
__global__ void k_htrans_b(const _Float16* __restrict__ in, _Float16* __restrict__ out,
                           int R, int C) {
  __shared__ _Float16 tile[32][33];
  int b = blockIdx.z;
  int rx = blockIdx.x * 32, cx = blockIdx.y * 32;
  const _Float16* ib = in + (size_t)b * R * C;
  _Float16* ob = out + (size_t)b * R * C;
  for (int r = threadIdx.y; r < 32; r += 8)
    tile[r][threadIdx.x] = ib[(size_t)(rx + r) * C + cx + threadIdx.x];
  __syncthreads();
  for (int r = threadIdx.y; r < 32; r += 8)
    ob[(size_t)(cx + r) * R + rx + threadIdx.x] = tile[threadIdx.x][r];
}

// ---------------------------------------------------------------- WMMA GEMM --
// C[M,N] = A[M,K] (f16 row-major) x Bt[N,K] (f16, pre-transposed).
// 8 waves/WG: 2 (M) x 4 (N); each wave computes a 16x64 strip.
__global__ void k_gemm_f16(const _Float16* __restrict__ A, const _Float16* __restrict__ Bt,
                           float* __restrict__ Cf, _Float16* __restrict__ Ch,
                           int M, int N, int K) {
  int lane = threadIdx.x & 31, wave = threadIdx.x >> 5;
  int lr = lane & 15, lh = lane >> 4;
  int m0 = blockIdx.x * 32 + (wave >> 2) * 16;
  int n0 = blockIdx.y * 256 + (wave & 3) * 64;
  const _Float16* arow = A + (size_t)(m0 + lr) * K + (lh << 3);
  v8f acc0 = {}, acc1 = {}, acc2 = {}, acc3 = {};
  for (int k0 = 0; k0 < K; k0 += 32) {
    v16h a;
    v8h alo = *(const v8h*)(arow + k0);
    v8h ahi = *(const v8h*)(arow + k0 + 16);
#pragma unroll
    for (int q = 0; q < 8; q++) { a[q] = alo[q]; a[8 + q] = ahi[q]; }
    int kb = k0 + (lh << 3);
    auto ldb = [&](const _Float16* p) { v8h l = *(const v8h*)p, u = *(const v8h*)(p + 16); v16h r; for (int q = 0; q < 8; q++) { r[q] = l[q]; r[8 + q] = u[q]; } return r; };
    v16h b0 = ldb(Bt + (size_t)(n0 +  0 + lr) * K + kb);
    v16h b1 = ldb(Bt + (size_t)(n0 + 16 + lr) * K + kb);
    v16h b2 = ldb(Bt + (size_t)(n0 + 32 + lr) * K + kb);
    v16h b3 = ldb(Bt + (size_t)(n0 + 48 + lr) * K + kb);
    acc0 = wmma16(a, b0, acc0);
    acc1 = wmma16(a, b1, acc1);
    acc2 = wmma16(a, b2, acc2);
    acc3 = wmma16(a, b3, acc3);
  }
#pragma unroll
  for (int v = 0; v < 8; v++) {
    int m = m0 + v + (lh << 3);
    size_t base = (size_t)m * N + n0 + lr;
    if (Cf) {
      Cf[base] = acc0[v]; Cf[base + 16] = acc1[v];
      Cf[base + 32] = acc2[v]; Cf[base + 48] = acc3[v];
    }
    if (Ch) {
      Ch[base] = (_Float16)acc0[v]; Ch[base + 16] = (_Float16)acc1[v];
      Ch[base + 32] = (_Float16)acc2[v]; Ch[base + 48] = (_Float16)acc3[v];
    }
  }
}

// --------------------------------------------------- scores + softmax fused --
// grid (L/16, NH, B), block 256 (8 waves). s = full 16x1024 score strip in LDS.
__global__ void k_attn(const _Float16* __restrict__ Qh, const _Float16* __restrict__ Kh,
                       float* __restrict__ attn) {
  __shared__ float s[16][1024];
  int i0 = blockIdx.x * 16, h = blockIdx.y, b = blockIdx.z;
  int lane = threadIdx.x & 31, wave = threadIdx.x >> 5;
  int lr = lane & 15, lh = lane >> 4;
  const _Float16* qrow = Qh + (size_t)(b * L_ + i0 + lr) * D_ + h * HD + (lh << 3);
  v16h a0, a1;
  {
    v8h lo0 = *(const v8h*)(qrow);      v8h hi0 = *(const v8h*)(qrow + 16);
    v8h lo1 = *(const v8h*)(qrow + 32); v8h hi1 = *(const v8h*)(qrow + 48);
#pragma unroll
    for (int q = 0; q < 8; q++) {
      a0[q] = lo0[q]; a0[8 + q] = hi0[q];
      a1[q] = lo1[q]; a1[8 + q] = hi1[q];
    }
  }
  for (int jt = wave; jt < 64; jt += 8) {
    const _Float16* krow = Kh + (size_t)(b * L_ + jt * 16 + lr) * D_ + h * HD + (lh << 3);
    v16h b0, b1;
    for (int q = 0; q < 8; q++) { b0[q] = krow[q]; b0[8 + q] = krow[16 + q]; b1[q] = krow[32 + q]; b1[8 + q] = krow[48 + q]; }
    v8f acc = {};
    acc = wmma16(a0, b0, acc);
    acc = wmma16(a1, b1, acc);
#pragma unroll
    for (int v = 0; v < 8; v++) s[v + (lh << 3)][jt * 16 + lr] = acc[v] * ATTN_SCALE;
  }
  __syncthreads();
  // softmax: 16 threads per row; 16-lane butterfly (xor masks 1/2/4/8 stay in-half, wave32)
  int r = threadIdx.x >> 4, c = threadIdx.x & 15;
  float mx = -3.0e38f;
  for (int k = 0; k < 64; k++) mx = fmaxf(mx, s[r][c + (k << 4)]);
  for (int mk = 1; mk < 16; mk <<= 1) mx = fmaxf(mx, __shfl_xor(mx, mk));
  float sum = 0.f;
  for (int k = 0; k < 64; k++) {
    float e = __expf(s[r][c + (k << 4)] - mx);
    s[r][c + (k << 4)] = e;
    sum += e;
  }
  for (int mk = 1; mk < 16; mk <<= 1) sum += __shfl_xor(sum, mk);
  float inv = 1.0f / sum;
  float* orow = attn + ((size_t)(b * NH + h) * L_ + i0 + r) * L_;
  for (int k = 0; k < 64; k++) orow[c + (k << 4)] = s[r][c + (k << 4)] * inv;
}

// ------------------------------------------------------------- relations ----
// grid (L/16, L/16, B), block 512 (16 waves; wave = rel index r).
// LDS staging [i][j][r] so the bijr output is written as contiguous 1KB rows.
__global__ void k_rel(const _Float16* __restrict__ Qrh, const _Float16* __restrict__ Krh,
                      float* __restrict__ rel) {
  __shared__ float rs[16 * 16 * 16];
  int i0 = blockIdx.x * 16, j0 = blockIdx.y * 16, b = blockIdx.z;
  int lane = threadIdx.x & 31, r = threadIdx.x >> 5;
  int lr = lane & 15, lh = lane >> 4;
  const _Float16* qrow = Qrh + (size_t)(b * L_ + i0 + lr) * D_ + r * RD + (lh << 3);
  const _Float16* krow = Krh + (size_t)(b * L_ + j0 + lr) * D_ + r * RD + (lh << 3);
  v16h a0, a1;
  {
    v8h lo0 = *(const v8h*)(qrow);      v8h hi0 = *(const v8h*)(qrow + 16);
    v8h lo1 = *(const v8h*)(qrow + 32); v8h hi1 = *(const v8h*)(qrow + 48);
#pragma unroll
    for (int q = 0; q < 8; q++) {
      a0[q] = lo0[q]; a0[8 + q] = hi0[q];
      a1[q] = lo1[q]; a1[8 + q] = hi1[q];
    }
  }
  v16h b0, b1;
  for (int q = 0; q < 8; q++) { b0[q] = krow[q]; b0[8 + q] = krow[16 + q]; b1[q] = krow[32 + q]; b1[8 + q] = krow[48 + q]; }
  v8f acc = {};
  acc = wmma16(a0, b0, acc);
  acc = wmma16(a1, b1, acc);
#pragma unroll
  for (int v = 0; v < 8; v++)
    rs[(v + (lh << 3)) * 256 + lr * 16 + r] = acc[v] * REL_SCALE;
  __syncthreads();
  int ib = threadIdx.x >> 8, e = threadIdx.x & 255;
#pragma unroll
  for (int ii = 0; ii < 8; ii++) {
    int i = ii * 2 + ib;
    rel[((size_t)(b * L_ + i0 + i) * L_ + j0) * 16 + e] = rs[i * 256 + e];
  }
}

// --------------------------------------------------- attended symbols -------
// grid (L/16, NH, B), block 128 (4 waves; wave = 16-wide d sub-tile).
__global__ void k_attnsym(const float* __restrict__ attn, const _Float16* __restrict__ SVt,
                          float* __restrict__ Y) {
  int i0 = blockIdx.x * 16, h = blockIdx.y, b = blockIdx.z;
  int lane = threadIdx.x & 31, w = threadIdx.x >> 5;
  int lr = lane & 15, lh = lane >> 4;
  const float* arow = attn + ((size_t)(b * NH + h) * L_ + i0 + lr) * L_ + (lh << 3);
  const _Float16* brow = SVt + ((size_t)b * D_ + h * HD + w * 16 + lr) * L_ + (lh << 3);
  v8f acc = {};
  for (int k0 = 0; k0 < L_; k0 += 32) {
    v16h a, fb;
#pragma unroll
    for (int q = 0; q < 8; q++) {
      a[q]     = (_Float16)arow[k0 + q];
      a[8 + q] = (_Float16)arow[k0 + 16 + q];
    }
    for (int q = 0; q < 8; q++) { fb[q] = brow[k0 + q]; fb[8 + q] = brow[k0 + 16 + q]; }
    acc = wmma16(a, fb, acc);
  }
#pragma unroll
  for (int v = 0; v < 8; v++)
    Y[(size_t)(b * L_ + i0 + v + (lh << 3)) * D_ + h * HD + w * 16 + lr] = acc[v];
}

// ---------------------------- t = attn x relations, then wr contraction -----
// grid (L, B), block 32 (one wave). Per (b,i): t[16h x 16r] via 32 WMMA steps,
// then Y[b,i,h,d] += sum_r t[h][r] * wr[h][d][r].
__global__ void k_rel_attend(const float* __restrict__ attn, const float* __restrict__ rel,
                             const float* __restrict__ wr, float* __restrict__ Y) {
  __shared__ float ts[16][16];
  int i = blockIdx.x, b = blockIdx.y;
  int lane = threadIdx.x;
  int lr = lane & 15, lh = lane >> 4;
  const float* arow = attn + ((size_t)(b * NH + lr) * L_ + i) * L_ + (lh << 3);
  const float* rbase = rel + (size_t)(b * L_ + i) * L_ * 16 + lr;
  v8f acc = {};
  for (int k0 = 0; k0 < L_; k0 += 32) {
    v16h a, fb;
#pragma unroll
    for (int q = 0; q < 8; q++) {
      a[q]     = (_Float16)arow[k0 + q];
      a[8 + q] = (_Float16)arow[k0 + 16 + q];
    }
    int jb = k0 + (lh << 3);
#pragma unroll
    for (int q = 0; q < 16; q++) fb[q] = (_Float16)rbase[(size_t)(jb + q + (q >> 3) * 8) * 16];
    acc = wmma16(a, fb, acc);
  }
#pragma unroll
  for (int v = 0; v < 8; v++) ts[v + (lh << 3)][lr] = acc[v];
  __syncthreads();
  for (int h = 0; h < 16; h++) {
#pragma unroll
    for (int dd = 0; dd < 2; dd++) {
      int d = lane + dd * 32;
      float sum = 0.f;
#pragma unroll
      for (int rr = 0; rr < 16; rr++) sum += ts[h][rr] * wr[((h << 6) + d) * 16 + rr];
      Y[(size_t)(b * L_ + i) * D_ + (h << 6) + d] += sum;
    }
  }
}

// ----------------------------------------------------------------- launch ---
extern "C" void kernel_launch(void* const* d_in, const int* in_sizes, int n_in,
                              void* d_out, int out_size, void* d_ws, size_t ws_size,
                              hipStream_t stream) {
  const float* x       = (const float*)d_in[0];
  const float* symbols = (const float*)d_in[1];
  const float* Wq_attn = (const float*)d_in[2];
  const float* Wk_attn = (const float*)d_in[3];
  const float* Wq_rel  = (const float*)d_in[4];
  const float* Wk_rel  = (const float*)d_in[5];
  const float* wr      = (const float*)d_in[6];
  const float* Wv      = (const float*)d_in[7];
  const float* Wo      = (const float*)d_in[8];

  float* out     = (float*)d_out;                       // [4,1024,1024]
  float* attnOut = out + (size_t)B_ * L_ * D_;          // [4,16,1024,1024]
  float* relOut  = attnOut + (size_t)B_ * NH * L_ * L_; // [4,1024,1024,16]

  // ---- workspace carve-up (≈100 MB) ----
  char* ws = (char*)d_ws;
  const size_t ACT_H = (size_t)BL * D_ * sizeof(_Float16); // 8 MB
  const size_t W_H   = (size_t)D_ * D_ * sizeof(_Float16); // 2 MB
  _Float16* Xh   = (_Float16*)ws;            ws += ACT_H;
  _Float16* Sh   = (_Float16*)ws;            ws += ACT_H;
  _Float16* WqaT = (_Float16*)ws;            ws += W_H;
  _Float16* WkaT = (_Float16*)ws;            ws += W_H;
  _Float16* WqrT = (_Float16*)ws;            ws += W_H;
  _Float16* WkrT = (_Float16*)ws;            ws += W_H;
  _Float16* WvT  = (_Float16*)ws;            ws += W_H;
  _Float16* WoT  = (_Float16*)ws;            ws += W_H;
  _Float16* Qh   = (_Float16*)ws;            ws += ACT_H;
  _Float16* Kh   = (_Float16*)ws;            ws += ACT_H;
  _Float16* Qrh  = (_Float16*)ws;            ws += ACT_H;
  _Float16* Krh  = (_Float16*)ws;            ws += ACT_H;
  _Float16* SVh  = (_Float16*)ws;            ws += ACT_H;
  _Float16* SVt  = (_Float16*)ws;            ws += ACT_H;
  _Float16* Yh   = (_Float16*)ws;            ws += ACT_H;
  float*    Y    = (float*)ws;               ws += (size_t)BL * D_ * sizeof(float);

  const int NACT = BL * D_;
  dim3 tb(32, 8), tw(32, 32), twb(32, 32, B_);
  dim3 gg(BL / 32, D_ / 256);

  // 1) converts + weight transposes
  k_cvt_f16<<<2048, 256, 0, stream>>>(x, Xh, NACT);
  k_cvt_f16<<<2048, 256, 0, stream>>>(symbols, Sh, NACT);
  k_wtrans<<<tw, tb, 0, stream>>>(Wq_attn, WqaT, D_, D_);
  k_wtrans<<<tw, tb, 0, stream>>>(Wk_attn, WkaT, D_, D_);
  k_wtrans<<<tw, tb, 0, stream>>>(Wq_rel,  WqrT, D_, D_);
  k_wtrans<<<tw, tb, 0, stream>>>(Wk_rel,  WkrT, D_, D_);
  k_wtrans<<<tw, tb, 0, stream>>>(Wv,      WvT,  D_, D_);
  k_wtrans<<<tw, tb, 0, stream>>>(Wo,      WoT,  D_, D_);

  // 2) projections (f16 outputs)
  k_gemm_f16<<<gg, 256, 0, stream>>>(Xh, WqaT, nullptr, Qh,  BL, D_, D_);
  k_gemm_f16<<<gg, 256, 0, stream>>>(Xh, WkaT, nullptr, Kh,  BL, D_, D_);
  k_gemm_f16<<<gg, 256, 0, stream>>>(Xh, WqrT, nullptr, Qrh, BL, D_, D_);
  k_gemm_f16<<<gg, 256, 0, stream>>>(Xh, WkrT, nullptr, Krh, BL, D_, D_);
  k_gemm_f16<<<gg, 256, 0, stream>>>(Sh, WvT,  nullptr, SVh, BL, D_, D_);
  k_htrans_b<<<twb, tb, 0, stream>>>(SVh, SVt, L_, D_);

  // 3) attn = softmax(QK^T * scale)  -> d_out region 2
  k_attn<<<dim3(L_ / 16, NH, B_), 256, 0, stream>>>(Qh, Kh, attnOut);

  // 4) relations = QrKr^T * scale    -> d_out region 3
  k_rel<<<dim3(L_ / 16, L_ / 16, B_), 512, 0, stream>>>(Qrh, Krh, relOut);

  // 5) Y = attn @ sv ; 6) Y += (attn*relations) @ wr
  k_attnsym<<<dim3(L_ / 16, NH, B_), 128, 0, stream>>>(attnOut, SVt, Y);
  k_rel_attend<<<dim3(L_, B_), 32, 0, stream>>>(attnOut, relOut, wr, Y);

  // 7) out = Y @ Wo
  k_cvt_f16<<<2048, 256, 0, stream>>>(Y, Yh, NACT);
  k_gemm_f16<<<gg, 256, 0, stream>>>(Yh, WoT, out, nullptr, BL, D_, D_);
}
